// VQEmbedding_11209864642757
// MI455X (gfx1250) — matrix operand order repaired and checked by reading, not compiled, not run on hardware
//
#include <hip/hip_runtime.h>

// CDNA5 / gfx1250: wave32, WMMA f32<-f16 16x16x32.
typedef __attribute__((ext_vector_type(16))) _Float16 v16h;
typedef __attribute__((ext_vector_type(8)))  float    v8f;
typedef __attribute__((ext_vector_type(4)))  float    v4f;   // native vector: OK for nontemporal builtins

#define DDIM   64
#define KCODES 512
#define CBS    72      // padded LDS row stride (halfs): 36 dwords, gcd(36,64)=4 -> good bank spread
#define ROWS_PER_BLOCK 256   // 8 waves * 2 strips * 16 rows

__global__ __launch_bounds__(256, 1)
void vq_wmma_kernel(const float* __restrict__ x,
                    const float* __restrict__ cb,
                    float* __restrict__ zq,
                    float* __restrict__ zqbar,
                    float* __restrict__ idxf)
{
    __shared__ _Float16 s_cb[KCODES * CBS];   // 72 KB: codebook in f16
    __shared__ float    s_csqr[KCODES];       // ||c||^2 in f32 (matches reference precision)
    __shared__ int      s_idx[8 * 32];        // per-wave argmin results (32 rows/wave)

    const int tid = threadIdx.x;

    for (int i = tid; i < KCODES; i += 256) s_csqr[i] = 0.0f;
    __syncthreads();

    // Fill LDS codebook (f32 -> f16) and accumulate ||c||^2 via ds_add_f32 atomics.
    for (int i = tid; i < KCODES * DDIM / 4; i += 256) {
        v4f v = ((const v4f*)cb)[i];
        int row = i >> 4;          // 16 float4 per code row
        int ch  = i & 15;
        union { _Float16 h[4]; uint2 u; } pk;
        pk.h[0] = (_Float16)v.x; pk.h[1] = (_Float16)v.y;
        pk.h[2] = (_Float16)v.z; pk.h[3] = (_Float16)v.w;
        *(uint2*)&s_cb[row * CBS + ch * 4] = pk.u;
        atomicAdd(&s_csqr[row], v.x*v.x + v.y*v.y + v.z*v.z + v.w*v.w);
    }
    __syncthreads();

    const int wave = tid >> 5;      // wave32
    const int lane = tid & 31;
    const int half = lane >> 4;     // 0: lanes 0-15, 1: lanes 16-31
    const int l    = lane & 15;

    const long rowbase = (long)blockIdx.x * ROWS_PER_BLOCK + (long)wave * 32;

    // ---- A fragments, two 16-row strips per wave. 16-bit A 16x32 layout:
    // lanes 0-15 (row l): VGPR0-3 = K0..7, VGPR4-7 = K16..23
    // lanes 16-31 (row l): VGPR0-3 = K8..15, VGPR4-7 = K24..31
    v16h A[2][2];    // [strip][k-slab]
    #pragma unroll
    for (int st = 0; st < 2; ++st) {
        const float* xr = x + (rowbase + st * 16 + l) * DDIM;
        #pragma unroll
        for (int s = 0; s < 2; ++s) {
            const int base = s * 32 + half * 8;
            union { v4f f4[4]; float f[16]; } a;
            a.f4[0] = *(const v4f*)(xr + base + 0);
            a.f4[1] = *(const v4f*)(xr + base + 4);
            a.f4[2] = *(const v4f*)(xr + base + 16);
            a.f4[3] = *(const v4f*)(xr + base + 20);
            #pragma unroll
            for (int i = 0; i < 16; ++i) A[st][s][i] = (_Float16)a.f[i];
        }
    }

    // Running per-lane (min dist, argmin). ||x||^2 dropped: constant per row.
    float best[2][8]; int bidx[2][8];
    #pragma unroll
    for (int st = 0; st < 2; ++st)
        #pragma unroll
        for (int r = 0; r < 8; ++r) { best[st][r] = 3.4e38f; bidx[st][r] = 0; }

    const _Float16* brBase = s_cb + l * CBS + half * 8;

    for (int j0 = 0; j0 < KCODES; j0 += 16) {
        // B fragments: dense 16-bit B 32x16 layout, shared by both row strips.
        // lane (half,l): column j0+l, K = half*8..+7 and 16+half*8..+7 (slab0), +32 (slab1)
        const _Float16* br = brBase + j0 * CBS;
        union { uint4 q[2]; v16h v; } B0, B1;
        B0.q[0] = *(const uint4*)(br + 0);
        B0.q[1] = *(const uint4*)(br + 16);
        B1.q[0] = *(const uint4*)(br + 32);
        B1.q[1] = *(const uint4*)(br + 48);

        // Two independent WMMA chains -> scheduler can cover XDL->VALU hazards.
        v8f acc0 = {}, acc1 = {};
        acc0 = __builtin_amdgcn_wmma_f32_16x16x32_f16(false, A[0][0], false, B0.v,
                                                      (short)0, acc0, false, false);
        acc1 = __builtin_amdgcn_wmma_f32_16x16x32_f16(false, A[1][0], false, B0.v,
                                                      (short)0, acc1, false, false);
        acc0 = __builtin_amdgcn_wmma_f32_16x16x32_f16(false, A[0][1], false, B1.v,
                                                      (short)0, acc0, false, false);
        acc1 = __builtin_amdgcn_wmma_f32_16x16x32_f16(false, A[1][1], false, B1.v,
                                                      (short)0, acc1, false, false);

        const float cs = s_csqr[j0 + l];
        const int   j  = j0 + l;
        #pragma unroll
        for (int r = 0; r < 8; ++r) {
            float d0 = fmaf(-2.0f, acc0[r], cs);   // ||c||^2 - 2 x.c
            float d1 = fmaf(-2.0f, acc1[r], cs);
            if (d0 < best[0][r]) { best[0][r] = d0; bidx[0][r] = j; } // j ascending -> first-min kept
            if (d1 < best[1][r]) { best[1][r] = d1; bidx[1][r] = j; }
        }
    }

    // Argmin across the 16 column-lanes of each half (C layout: rows r / r+8).
    // Branch-free tie-break: smaller index wins on equal distance.
    #pragma unroll
    for (int m = 1; m < 16; m <<= 1) {
        #pragma unroll
        for (int st = 0; st < 2; ++st) {
            #pragma unroll
            for (int r = 0; r < 8; ++r) {
                float od = __shfl_xor(best[st][r], m, 32);
                int   oi = __shfl_xor(bidx[st][r], m, 32);
                bool take = (od < best[st][r]) |
                            ((od == best[st][r]) & (oi < bidx[st][r]));
                best[st][r] = take ? od : best[st][r];
                bidx[st][r] = take ? oi : bidx[st][r];
            }
        }
    }

    if (l == 0) {   // lane 0 -> rows 0..7 of strip, lane 16 -> rows 8..15
        #pragma unroll
        for (int st = 0; st < 2; ++st)
            #pragma unroll
            for (int r = 0; r < 8; ++r)
                s_idx[wave * 32 + st * 16 + half * 8 + r] = bidx[st][r];
    }
    __syncthreads();

    // indices output (d_out is float)
    idxf[rowbase + lane] = (float)s_idx[wave * 32 + lane];

    // Gather from ORIGINAL f32 codebook (L2-hot) and stream both outputs.
    // 32 rows * 16 float4 = 512 tasks -> 16 per lane.
    #pragma unroll
    for (int t = 0; t < 16; ++t) {
        int flat = t * 32 + lane;
        int m = flat >> 4, c = flat & 15;
        int code = s_idx[wave * 32 + m];
        v4f v = *(const v4f*)(cb + code * DDIM + c * 4);
        long o = (rowbase + m) * DDIM + c * 4;
        __builtin_nontemporal_store(v, (v4f*)(zq + o));
        __builtin_nontemporal_store(v, (v4f*)(zqbar + o));
    }
}

extern "C" void kernel_launch(void* const* d_in, const int* in_sizes, int n_in,
                              void* d_out, int out_size, void* d_ws, size_t ws_size,
                              hipStream_t stream) {
    const float* x  = (const float*)d_in[0];   // z_e_x  [N,64] f32
    const float* cb = (const float*)d_in[1];   // codebook [512,64] f32
    const long ND = in_sizes[0];               // N*64
    const long n  = ND / DDIM;                 // N rows

    float* out   = (float*)d_out;
    float* zq    = out;                        // [N,64]
    float* zqbar = out + ND;                   // [N,64]
    float* idxf  = out + 2 * ND;               // [N] indices as float

    const int blocks = (int)(n / ROWS_PER_BLOCK);   // N=262144 -> 1024, exact
    vq_wmma_kernel<<<blocks, 256, 0, stream>>>(x, cb, zq, zqbar, idxf);
}
